// BahdanauAttention_15874199126101
// MI455X (gfx1250) — matrix operand order repaired and checked by reading, not compiled, not run on hardware
//
#include <hip/hip_runtime.h>
#include <hip/hip_bf16.h>
#include <math.h>

// ---------------------------------------------------------------------------
// Types for CDNA5 WMMA (wave32): 16x16x32 bf16 -> f32 accum
//   A/B operand: 16 bf16 per lane (8 VGPRs), C/D: 8 f32 per lane (8 VGPRs)
// ---------------------------------------------------------------------------
typedef __attribute__((ext_vector_type(16))) __bf16 v16bf;
typedef __attribute__((ext_vector_type(8)))  float  v8f;

#define B_DIM   16
#define T_Q     64
#define T_K     128
#define N_UNITS 1024
#define K_DEPTH 1024

// ---------------------------------------------------------------------------
// Kernel A: la_scaled[n] = linear_att[n] / ||linear_att|| * normalize_scalar
// ---------------------------------------------------------------------------
__global__ void la_normalize_kernel(const float* __restrict__ la_in,
                                    const float* __restrict__ nscalar,
                                    float* __restrict__ la_out) {
    __shared__ float red[8];
    __shared__ float s_scale;
    const int tid  = threadIdx.x;
    const int lane = tid & 31;
    const int wave = tid >> 5;

    float4 v = *(const float4*)(la_in + tid * 4);
    float s = v.x * v.x + v.y * v.y + v.z * v.z + v.w * v.w;
    #pragma unroll
    for (int off = 16; off >= 1; off >>= 1) s += __shfl_xor(s, off, 32);
    if (lane == 0) red[wave] = s;
    __syncthreads();
    if (tid == 0) {
        float t = 0.f;
        #pragma unroll
        for (int i = 0; i < 8; ++i) t += red[i];
        s_scale = nscalar[0] * rsqrtf(t);
    }
    __syncthreads();
    const float sc = s_scale;
    float4 o; o.x = v.x * sc; o.y = v.y * sc; o.z = v.z * sc; o.w = v.w * sc;
    *(float4*)(la_out + tid * 4) = o;
}

// ---------------------------------------------------------------------------
// f32 -> bf16 operand packing helpers (per-lane contiguous 32B-aligned loads)
// ---------------------------------------------------------------------------
__device__ __forceinline__ void cvt8(const float* __restrict__ p, v16bf& d, int base) {
    float4 x = *(const float4*)(p);
    float4 y = *(const float4*)(p + 4);
    d[base + 0] = (__bf16)x.x; d[base + 1] = (__bf16)x.y;
    d[base + 2] = (__bf16)x.z; d[base + 3] = (__bf16)x.w;
    d[base + 4] = (__bf16)y.x; d[base + 5] = (__bf16)y.y;
    d[base + 6] = (__bf16)y.z; d[base + 7] = (__bf16)y.w;
}

// Build one K=32 slice of the A operand for this lane.
// bf16 A layout: lane<16 -> K {kk..kk+7, kk+16..kk+23}
//                lane>=16 -> K {kk+8..kk+15, kk+24..kk+31}
__device__ __forceinline__ v16bf load_a_slice(const float* __restrict__ arow,
                                              int kk, int hi) {
    v16bf a;
    const float* ap = arow + kk + hi * 8;
    cvt8(ap,      a, 0);
    cvt8(ap + 16, a, 8);
    return a;
}

// Build one K=32 slice of the B operand (B[k][n] = W[n][k], lane = n column).
// bf16 B layout: same as A: lane<16 -> K {kk..kk+7, kk+16..kk+23}, lane>=16 -> K {kk+8..kk+15, kk+24..kk+31}.
__device__ __forceinline__ v16bf load_b_slice(const float* __restrict__ wrow,
                                              int kk, int hi) {
    v16bf b;
    cvt8(wrow + kk + hi * 8,      b, 0);
    cvt8(wrow + kk + hi * 8 + 16, b, 8);
    return b;
}

// ---------------------------------------------------------------------------
// Kernel B: projection GEMM  out[r,n] = sum_i A[r,i] * W[n,i] (+ bias[n])
// A row-major (M x 1024), W row-major (1024 x 1024) (nn.Linear layout).
// One wave per 16x16 output tile; K unrolled x2 -> two dependent (shared c)
// back-to-back v_wmma_f32_16x16x32_bf16 per iteration.
// N, K fixed at 1024 so all addressing is shift+add; HAS_BIAS templated so
// the epilogue is branch-free.
// ---------------------------------------------------------------------------
template <bool HAS_BIAS>
__global__ void wmma_proj_gemm(const float* __restrict__ A,
                               const float* __restrict__ W,
                               const float* __restrict__ bias,
                               float* __restrict__ out,
                               int totalTiles) {
    constexpr int N = N_UNITS;
    constexpr int K = K_DEPTH;
    constexpr int tilesN = N / 16;            // 64

    const int lane = threadIdx.x & 31;
    const int wave = threadIdx.x >> 5;
    const int tile = blockIdx.x * (blockDim.x >> 5) + wave;
    if (tile >= totalTiles) return;           // wave-uniform: EXEC stays all-1s

    const int tm  = tile >> 6;                // tile / tilesN
    const int tn  = tile & (tilesN - 1);
    const int hi  = lane >> 4;                // half-wave select
    const int l16 = lane & 15;

    const float* arow = A + (size_t)(tm * 16 + l16) * K;  // A lane = out row
    const float* wrow = W + (size_t)(tn * 16 + l16) * K;  // B lane = out col

    v8f c = {};
    for (int kk = 0; kk < K; kk += 64) {
        v16bf a0 = load_a_slice(arow, kk,      hi);
        v16bf b0 = load_b_slice(wrow, kk,      hi);
        v16bf a1 = load_a_slice(arow, kk + 32, hi);
        v16bf b1 = load_b_slice(wrow, kk + 32, hi);
        c = __builtin_amdgcn_wmma_f32_16x16x32_bf16(
                false, a0, false, b0, (short)0, c, false, false);
        c = __builtin_amdgcn_wmma_f32_16x16x32_bf16(
                false, a1, false, b1, (short)0, c, false, false);
    }

    // C/D layout: VGPR j -> M = j + 8*hi, N = lane&15
    const int col   = tn * 16 + l16;
    const int rbase = tm * 16 + hi * 8;
    float bv = 0.0f;
    if (HAS_BIAS) bv = bias[col];
    float* orow = out + (size_t)rbase * N + col;
    #pragma unroll
    for (int j = 0; j < 8; ++j)
        orow[(size_t)j * N] = c[j] + bv;
}

// ---------------------------------------------------------------------------
// Kernel C: fused scores + softmax + context. One block (256 thr) per (b,q).
//   scores[k] = sum_n tanh(aq[b,q,n] + ak[b,k,n]) * la[n]   (bias is in aq)
//   p = softmax_k(scores);  ctx[n] = sum_k p[k] * keys[b,k,n]
// aq row (1024 f) and la cached in registers (32/lane); each wave owns 16 k's.
// Never materializes the (b,q,k,n) tensor (saves ~1 GB of memory traffic).
// ---------------------------------------------------------------------------
__global__ void attn_fused_kernel(const float* __restrict__ aq,
                                  const float* __restrict__ ak,
                                  const float* __restrict__ keys,
                                  const float* __restrict__ la,
                                  float* __restrict__ out_ctx,
                                  float* __restrict__ out_scores) {
    __shared__ float s_p[T_K];
    const int bq   = blockIdx.x;       // b*T_Q + q
    const int b    = bq >> 6;
    const int tid  = threadIdx.x;
    const int lane = tid & 31;
    const int wave = tid >> 5;

    // cache this (b,q)'s aq row and la in registers: 32 floats each per lane
    float aql[32], lal[32];
    {
        const float* aqp = aq + (size_t)bq * N_UNITS + lane * 32;
        const float* lap = la + lane * 32;
        #pragma unroll
        for (int i = 0; i < 32; i += 4) {
            float4 t = *(const float4*)(aqp + i);
            aql[i] = t.x; aql[i + 1] = t.y; aql[i + 2] = t.z; aql[i + 3] = t.w;
            float4 u = *(const float4*)(lap + i);
            lal[i] = u.x; lal[i + 1] = u.y; lal[i + 2] = u.z; lal[i + 3] = u.w;
        }
    }

    // phase 1: each wave computes 16 of the 128 scores
    for (int kk = 0; kk < 16; ++kk) {
        const int k = wave + kk * 8;
        const float* akp = ak + ((size_t)(b * T_K + k)) * N_UNITS + lane * 32;
        float acc = 0.f;
        #pragma unroll
        for (int i = 0; i < 32; i += 4) {
            float4 t = *(const float4*)(akp + i);
            acc += tanhf(aql[i + 0] + t.x) * lal[i + 0];
            acc += tanhf(aql[i + 1] + t.y) * lal[i + 1];
            acc += tanhf(aql[i + 2] + t.z) * lal[i + 2];
            acc += tanhf(aql[i + 3] + t.w) * lal[i + 3];
        }
        #pragma unroll
        for (int off = 16; off >= 1; off >>= 1) acc += __shfl_xor(acc, off, 32);
        if (lane == 0) s_p[k] = acc;
    }
    __syncthreads();

    // phase 2: softmax over k=128 (wave 0 only; 4 values per lane)
    if (wave == 0) {
        float v0 = s_p[lane], v1 = s_p[lane + 32],
              v2 = s_p[lane + 64], v3 = s_p[lane + 96];
        float m = fmaxf(fmaxf(v0, v1), fmaxf(v2, v3));
        #pragma unroll
        for (int off = 16; off >= 1; off >>= 1)
            m = fmaxf(m, __shfl_xor(m, off, 32));
        float e0 = __expf(v0 - m), e1 = __expf(v1 - m),
              e2 = __expf(v2 - m), e3 = __expf(v3 - m);
        float sum = e0 + e1 + e2 + e3;
        #pragma unroll
        for (int off = 16; off >= 1; off >>= 1) sum += __shfl_xor(sum, off, 32);
        const float inv = 1.0f / sum;
        s_p[lane]      = e0 * inv;
        s_p[lane + 32] = e1 * inv;
        s_p[lane + 64] = e2 * inv;
        s_p[lane + 96] = e3 * inv;
    }
    __syncthreads();

    if (tid < T_K) out_scores[(size_t)bq * T_K + tid] = s_p[tid];

    // phase 3: context GEMV  ctx[n] = sum_k p[k]*keys[b,k,n]; 4 n per thread
    float4 acc4 = make_float4(0.f, 0.f, 0.f, 0.f);
    const float* kb = keys + (size_t)b * T_K * N_UNITS + tid * 4;
    #pragma unroll 4
    for (int k = 0; k < T_K; ++k) {
        const float pk = s_p[k];
        float4 kv = *(const float4*)(kb + (size_t)k * N_UNITS);
        acc4.x += pk * kv.x; acc4.y += pk * kv.y;
        acc4.z += pk * kv.z; acc4.w += pk * kv.w;
    }
    *(float4*)(out_ctx + (size_t)bq * N_UNITS + tid * 4) = acc4;
}

// ---------------------------------------------------------------------------
// launcher
// ---------------------------------------------------------------------------
extern "C" void kernel_launch(void* const* d_in, const int* in_sizes, int n_in,
                              void* d_out, int out_size, void* d_ws, size_t ws_size,
                              hipStream_t stream) {
    const float* query = (const float*)d_in[0];  // (16,64,1024)
    const float* keys  = (const float*)d_in[1];  // (16,128,1024)
    const float* Wq    = (const float*)d_in[2];  // (1024,1024)
    const float* Wk    = (const float*)d_in[3];  // (1024,1024)
    const float* la    = (const float*)d_in[4];  // (1024,)
    const float* nsc   = (const float*)d_in[5];  // (1,)
    const float* nbias = (const float*)d_in[6];  // (1024,)

    float* out_ctx    = (float*)d_out;                                 // (16,64,1024)
    float* out_scores = (float*)d_out + (size_t)B_DIM * T_Q * N_UNITS; // (16,64,128)

    float* ws   = (float*)d_ws;
    float* la_s = ws;                                  // 1024
    float* aq   = ws + 1024;                           // 16*64*1024
    float* ak   = aq + (size_t)B_DIM * T_Q * N_UNITS;  // 16*128*1024

    // 1) normalized attention vector
    la_normalize_kernel<<<1, 256, 0, stream>>>(la, nsc, la_s);

    // 2) aq = query·Wq^T + bias : (1024/16)*(1024/16) = 4096 tiles, 8 waves/blk
    wmma_proj_gemm<true><<<512, 256, 0, stream>>>(query, Wq, nbias, aq, 4096);
    // 3) ak = keys·Wk^T : (2048/16)*(1024/16) = 8192 tiles
    wmma_proj_gemm<false><<<1024, 256, 0, stream>>>(keys, Wk, nullptr, ak, 8192);

    // 4) fused tanh-scores + softmax + context, one block per (b,q)
    attn_fused_kernel<<<B_DIM * T_Q, 256, 0, stream>>>(aq, ak, keys, la_s,
                                                       out_ctx, out_scores);
}
